// EdgeModel_NoMem_26302379720743
// MI455X (gfx1250) — matrix operand order repaired and checked by reading, not compiled, not run on hardware
//
#include <hip/hip_runtime.h>

typedef _Float16 v16h __attribute__((ext_vector_type(16)));
typedef _Float16 v8h  __attribute__((ext_vector_type(8)));
typedef float    v8f  __attribute__((ext_vector_type(8)));

#define E_TOTAL 1000000
#define TILES   62500          // E / 16 (exact)
#define NBLK    1024
#define WPB     8              // waves per block (256 threads, wave32)

// sched_group_barrier masks
#define SG_MFMA    0x008      // MFMA/WMMA group
#define SG_DSREAD  0x100      // DS read group

// ---------------------------------------------------------------------------
// Per-wave 16-edge tile MLP using v_wmma_f32_16x16x32_f16.
//
// A-matrix f16 16x32 layout (ISA 7.12.2): lane -> row m = lane&15,
//   K values = {kb..kb+7} U {16+kb..16+kb+7}, kb = (lane>=16)*8.
// B-matrix f16 32x16 layout: lane -> col n = lane&15,
//   K = (h>>3)*16 + (lane>=16)*8 + (h&7) for half index h = 0..15.
// C/D f32 16x16 layout: lane -> col n = lane&15, VGPR r -> row m = r + (lane>=16)*8.
// ---------------------------------------------------------------------------

__device__ __forceinline__ v16h cvt_a(v8f f0, v8f f1) {
  v16h a;
#pragma unroll
  for (int i = 0; i < 8; ++i) {
    a[i]     = (_Float16)f0[i];
    a[8 + i] = (_Float16)f1[i];
  }
  return a;
}

__global__ __launch_bounds__(256) void edge_mlp_kernel(
    const float* __restrict__ src, const float* __restrict__ dest,
    const float* __restrict__ u, const long long* __restrict__ batch,
    const float* __restrict__ W1, const float* __restrict__ b1,
    const float* __restrict__ W2, const float* __restrict__ b2,
    float* __restrict__ out)
{
  // W1 fragments: 5 ktiles x 8 ntiles = 40 frags * 512 halfs  (40 KB)
  // W2 fragments: 4 ktiles x 4 ntiles = 16 frags * 512 halfs  (16 KB)
  // h scratch   : 8 waves * 16 rows * 128 cols f16            (32 KB)
  __shared__ _Float16 sW1[40 * 512];
  __shared__ _Float16 sW2[16 * 512];
  __shared__ _Float16 sH[WPB * 16 * 128];

  // ---- one-time per block: convert fp32 weights into f16 B-fragment layout
  for (int idx = threadIdx.x; idx < 40 * 512; idx += 256) {
    int frag = idx >> 9, rem = idx & 511;
    int lane = rem >> 4, h = rem & 15;
    int kt = frag >> 3, nt = frag & 7;
    int n = nt * 16 + (lane & 15);
    int k = kt * 32 + ((h >> 3) << 4) + ((lane >> 4) << 3) + (h & 7);   // K within [0,160)
    sW1[idx] = (_Float16)W1[k * 128 + n];
  }
  for (int idx = threadIdx.x; idx < 16 * 512; idx += 256) {
    int frag = idx >> 9, rem = idx & 511;
    int lane = rem >> 4, h = rem & 15;
    int kt = frag >> 2, nt = frag & 3;
    int n = nt * 16 + (lane & 15);
    int k = kt * 32 + ((h >> 3) << 4) + ((lane >> 4) << 3) + (h & 7);   // K within [0,128)
    sW2[idx] = (_Float16)W2[k * 64 + n];
  }
  __syncthreads();

  const int w    = threadIdx.x >> 5;
  const int lane = threadIdx.x & 31;
  const int ml   = lane & 15;
  const int hi   = lane >> 4;
  const int kb   = hi * 8;
  _Float16* __restrict__ hbuf = &sH[w * 2048];

  // biases per lane (N = nt*16 + ml in C/D layout), uniform over VGPR rows
  float b1v[8], b2v[4];
#pragma unroll
  for (int nt = 0; nt < 8; ++nt) b1v[nt] = b1[nt * 16 + ml];
#pragma unroll
  for (int nt = 0; nt < 4; ++nt) b2v[nt] = b2[nt * 16 + ml];

#pragma clang loop unroll(disable)
  for (int tile = blockIdx.x * WPB + w; tile < TILES; tile += NBLK * WPB) {
    // Prevent LICM from hoisting the 56 LDS B-fragment loads out of the
    // loop (would need ~450 VGPRs and spill).
    asm volatile("" ::: "memory");

    const long e = (long)tile * 16 + ml;            // edge row owned by lane
    const float* __restrict__ srow = src  + e * 64;
    const float* __restrict__ drow = dest + e * 64;
    const long long be = batch[e];
    const float* __restrict__ urow = u + (long)be * 32;

    // ---- A chunks (global loads + cvt float freely in the schedule) ----
    v16h afr[5];
    afr[0] = cvt_a(*(const v8f*)(srow + kb),      *(const v8f*)(srow + 16 + kb));
    afr[1] = cvt_a(*(const v8f*)(srow + 32 + kb), *(const v8f*)(srow + 48 + kb));
    afr[2] = cvt_a(*(const v8f*)(drow + kb),      *(const v8f*)(drow + 16 + kb));
    afr[3] = cvt_a(*(const v8f*)(drow + 32 + kb), *(const v8f*)(drow + 48 + kb));
    afr[4] = cvt_a(*(const v8f*)(urow + kb),      *(const v8f*)(urow + 16 + kb));

    // ---- layer 1: [16x160] @ [160x128] + b1 ----
    v8f acc[8];
#pragma unroll
    for (int nt = 0; nt < 8; ++nt) {
      v8f c;
#pragma unroll
      for (int r = 0; r < 8; ++r) c[r] = b1v[nt];
      acc[nt] = c;
    }

    // Two half-chains per kt: [8 ds reads][4 wmma][8 ds reads][4 wmma].
    // Keeps 4 B fragments (32 VGPRs) in flight -> partial dscnt waits
    // without pushing allocation past 256 VGPRs.
#pragma unroll
    for (int kt = 0; kt < 5; ++kt) {
#pragma unroll
      for (int half = 0; half < 2; ++half) {
        v16h bfr[4];
#pragma unroll
        for (int i = 0; i < 4; ++i)
          bfr[i] = *(const v16h*)
              &sW1[(kt * 8 + half * 4 + i) * 512 + lane * 16];
        __builtin_amdgcn_sched_group_barrier(SG_DSREAD, 8, 0);
#pragma unroll
        for (int i = 0; i < 4; ++i) {
          int nt = half * 4 + i;
          acc[nt] = __builtin_amdgcn_wmma_f32_16x16x32_f16(
              false, afr[kt], false, bfr[i], (short)0, acc[nt], false, false);
        }
        __builtin_amdgcn_sched_group_barrier(SG_MFMA, 4, 0);
      }
    }

    // ---- relu + f16, D-layout -> row-major LDS (transpose for layer-2 A) ----
#pragma unroll
    for (int nt = 0; nt < 8; ++nt) {
#pragma unroll
      for (int r = 0; r < 8; ++r) {
        float v = acc[nt][r];
        v = v > 0.0f ? v : 0.0f;
        hbuf[(hi * 8 + r) * 128 + nt * 16 + ml] = (_Float16)v;
      }
    }

    // ---- layer 2: [16x128] @ [128x64] + b2 ----
    // Preload all 4 h A-fragments (one store->load turnaround wait).
    v16h hfr[4];
#pragma unroll
    for (int kt = 0; kt < 4; ++kt) {
      const _Float16* hp = hbuf + ml * 128 + kt * 32 + kb;   // 16B aligned
      v8h lo  = *(const v8h*)hp;
      v8h hi8 = *(const v8h*)(hp + 16);
      hfr[kt] = __builtin_shufflevector(lo, hi8,
          0, 1, 2, 3, 4, 5, 6, 7, 8, 9, 10, 11, 12, 13, 14, 15);
    }
    __builtin_amdgcn_sched_group_barrier(SG_DSREAD, 8, 0);

    v8f acc2[4];
#pragma unroll
    for (int nt = 0; nt < 4; ++nt) {
      v8f c;
#pragma unroll
      for (int r = 0; r < 8; ++r) c[r] = b2v[nt];
      acc2[nt] = c;
    }

#pragma unroll
    for (int kt = 0; kt < 4; ++kt) {
      v16h bfr[4];
#pragma unroll
      for (int nt = 0; nt < 4; ++nt)
        bfr[nt] = *(const v16h*)&sW2[(kt * 4 + nt) * 512 + lane * 16];
      __builtin_amdgcn_sched_group_barrier(SG_DSREAD, 8, 0);
#pragma unroll
      for (int nt = 0; nt < 4; ++nt)
        acc2[nt] = __builtin_amdgcn_wmma_f32_16x16x32_f16(
            false, hfr[kt], false, bfr[nt], (short)0, acc2[nt], false, false);
      __builtin_amdgcn_sched_group_barrier(SG_MFMA, 4, 0);
    }

    // ---- store output: lane writes col n = nt*16+ml of row tile*16 + hi*8 + r
    float* __restrict__ orow = out + (long)tile * 16 * 64;
#pragma unroll
    for (int nt = 0; nt < 4; ++nt) {
#pragma unroll
      for (int r = 0; r < 8; ++r) {
        orow[(hi * 8 + r) * 64 + nt * 16 + ml] = acc2[nt][r];
      }
    }
  }
}

extern "C" void kernel_launch(void* const* d_in, const int* in_sizes, int n_in,
                              void* d_out, int out_size, void* d_ws, size_t ws_size,
                              hipStream_t stream) {
  const float*     src   = (const float*)d_in[0];
  const float*     dest  = (const float*)d_in[1];
  const float*     u     = (const float*)d_in[2];
  const long long* batch = (const long long*)d_in[3];
  const float*     W1    = (const float*)d_in[4];
  const float*     b1    = (const float*)d_in[5];
  const float*     W2    = (const float*)d_in[6];
  const float*     b2    = (const float*)d_in[7];
  float*           out   = (float*)d_out;

  edge_mlp_kernel<<<NBLK, 256, 0, stream>>>(src, dest, u, batch, W1, b1, W2, b2, out);
}
